// TransformerBlock_36369783063327
// MI455X (gfx1250) — matrix operand order repaired and checked by reading, not compiled, not run on hardware
//
#include <hip/hip_runtime.h>
#include <hip/hip_bf16.h>

// ---------------------------------------------------------------------------
// Transformer block for MI455X (gfx1250, wave32, WMMA + async-to-LDS).
// ---------------------------------------------------------------------------

typedef __attribute__((ext_vector_type(16))) __bf16 bf16x16;
typedef __attribute__((ext_vector_type(8)))  __bf16 bf16x8;
typedef __attribute__((ext_vector_type(8)))  float  f32x8;

#define D_MODEL 1024
#define HEADS   16
#define DK      64
#define SEQ     2048
#define BATCH   2
#define ROWS    (BATCH*SEQ)   /* 4096 */
#define D_FF    4096
#define LN_EPS  1e-6f

// ---------------------------------------------------------------------------
// CDNA5 async copy global->LDS (ASYNCcnt path). The generic pointer to
// __shared__ memory carries the LDS byte offset in its low 32 bits
// (ISA 10.2 aperture rules), which is exactly what the instruction's VDST
// LDS-address VGPR wants.
// ---------------------------------------------------------------------------
__device__ __forceinline__ void async_copy_b128(const void* gsrc, void* lds_dst) {
    uint32_t loff = (uint32_t)(uintptr_t)lds_dst;
    asm volatile("global_load_async_to_lds_b128 %0, %1, off"
                 :: "v"(loff), "v"(gsrc) : "memory");
}

template <int N>
__device__ __forceinline__ void wait_asynccnt() {
#if __has_builtin(__builtin_amdgcn_s_wait_asynccnt)
    __builtin_amdgcn_s_wait_asynccnt(N);
#else
    asm volatile("s_wait_asynccnt %0" :: "i"(N) : "memory");
#endif
}

// ---------------------------------------------------------------------------
// f32 -> bf16 conversion
// ---------------------------------------------------------------------------
__global__ void cvt_f32_bf16(const float* __restrict__ in,
                             __bf16* __restrict__ out, int n) {
    int i = blockIdx.x * blockDim.x + threadIdx.x;
    if (i < n) out[i] = (__bf16)in[i];
}

// ---------------------------------------------------------------------------
// C[M,N] = A[M,K] * Bw[N,K]^T   (bf16 in, f32 accumulate)
// Block = 256 threads = 8 waves (2 M x 4 N); wave tile 16M x 64N;
// block tile 32M x 256N. K staged through LDS in 64-deep double-buffered
// stages moved by global_load_async_to_lds_b128.
// EPI: 0 = store f32, 1 = store bf16, 2 = relu + store bf16
// ---------------------------------------------------------------------------
#define GEMM_KS     64
#define LDSA_ROWS   32
#define LDSB_ROWS   256
#define LDS_ROWS    (LDSA_ROWS + LDSB_ROWS)           /* 288 */
#define LDS_STRIDE  72   /* 64 data + 8 pad bf16; 144 B = 9 x 16 B chunks */
#define STAGE_CHUNKS_PER_THREAD 9   /* 288 rows * 8 chunks / 256 threads */

template <int EPI>
__global__ void __launch_bounds__(256)
gemm_bf16_nt(const __bf16* __restrict__ A, const __bf16* __restrict__ Bw,
             void* __restrict__ Cout, int M, int N, int K) {
    __shared__ __align__(16) __bf16 tile[2][LDS_ROWS * LDS_STRIDE];

    const int tid  = threadIdx.x;
    const int lane = tid & 31;
    const int wave = tid >> 5;
    const int hi   = lane >> 4;
    const int lid  = lane & 15;
    const int mBlk = blockIdx.y * 32;
    const int nBlk = blockIdx.x * 256;
    const int waveM = (wave & 1) * 16;     // row offset inside A tile
    const int waveN = (wave >> 1) * 64;    // row offset inside B tile

    // issue one 64-deep K stage of async copies (9 b128 chunks per thread)
    auto stage = [&](int k0, int buf) {
        {   // chunks 0..255: A tile (32 rows x 8 chunks)
            int row = tid >> 3, ch = tid & 7;
            async_copy_b128(A + (size_t)(mBlk + row) * K + k0 + ch * 8,
                            &tile[buf][row * LDS_STRIDE + ch * 8]);
        }
        #pragma unroll
        for (int t = 1; t < STAGE_CHUNKS_PER_THREAD; ++t) {
            // chunks 256..2303: B tile (256 rows x 8 chunks)
            int c   = tid + t * 256;
            int row = (c >> 3) - LDSA_ROWS;
            int ch  = c & 7;
            async_copy_b128(Bw + (size_t)(nBlk + row) * K + k0 + ch * 8,
                            &tile[buf][(LDSA_ROWS + row) * LDS_STRIDE + ch * 8]);
        }
    };

    f32x8 acc[4] = {};
    stage(0, 0);
    int cur = 0;

    for (int k0 = 0; k0 < K; k0 += GEMM_KS) {
        const bool more = (k0 + GEMM_KS) < K;
        if (more) {
            stage(k0 + GEMM_KS, cur ^ 1);          // next stage in flight
            wait_asynccnt<STAGE_CHUNKS_PER_THREAD>();  // current stage landed
        } else {
            wait_asynccnt<0>();
        }
        __syncthreads();

        const __bf16* at = &tile[cur][0];
        const __bf16* bt = &tile[cur][LDSA_ROWS * LDS_STRIDE];

        #pragma unroll
        for (int kk = 0; kk < GEMM_KS; kk += 32) {
            bf16x16 af;
            {
                const __bf16* ar = at + (waveM + lid) * LDS_STRIDE + kk;
                bf16x8 x1 = *(const bf16x8*)(ar + hi * 8);
                bf16x8 x2 = *(const bf16x8*)(ar + 16 + hi * 8);
                #pragma unroll
                for (int e = 0; e < 8; ++e) { af[e] = x1[e]; af[e + 8] = x2[e]; }
            }
            #pragma unroll
            for (int j = 0; j < 4; ++j) {
                const __bf16* br =
                    bt + (waveN + j * 16 + lid) * LDS_STRIDE + kk;
                bf16x8 y1 = *(const bf16x8*)(br + hi * 8);
                bf16x8 y2 = *(const bf16x8*)(br + 16 + hi * 8);
                bf16x16 bfr;
                #pragma unroll
                for (int e = 0; e < 8; ++e) { bfr[e] = y1[e]; bfr[e + 8] = y2[e]; }
                acc[j] = __builtin_amdgcn_wmma_f32_16x16x32_bf16(
                    false, af, false, bfr, (short)0, acc[j], false, false);
            }
        }
        __syncthreads();   // guards buf reuse before next issue overwrites it
        cur ^= 1;
    }

    #pragma unroll
    for (int j = 0; j < 4; ++j) {
        const int col = nBlk + waveN + j * 16 + lid;
        #pragma unroll
        for (int r = 0; r < 8; ++r) {
            const int row = mBlk + waveM + hi * 8 + r;
            float v = acc[j][r];
            if (EPI == 0) {
                ((float*)Cout)[(size_t)row * N + col] = v;
            } else if (EPI == 1) {
                ((__bf16*)Cout)[(size_t)row * N + col] = (__bf16)v;
            } else {
                ((__bf16*)Cout)[(size_t)row * N + col] =
                    (__bf16)(v > 0.f ? v : 0.f);
            }
        }
    }
}

// ---------------------------------------------------------------------------
// V [4096,1024] -> Vt [bh=32][dv=64][s=2048]  (bf16)
// ---------------------------------------------------------------------------
__global__ void transpose_v(const __bf16* __restrict__ V,
                            __bf16* __restrict__ Vt, int total) {
    int i = blockIdx.x * blockDim.x + threadIdx.x;
    if (i >= total) return;
    int s  = i & (SEQ - 1);
    int t  = i >> 11;            // bh*64 + dv
    int dv = t & 63;
    int bh = t >> 6;
    int b  = bh >> 4, h = bh & 15;
    Vt[i] = V[(size_t)(b * SEQ + s) * D_MODEL + h * DK + dv];
}

// ---------------------------------------------------------------------------
// Flash attention (causal). Grid: (SEQ/128, BATCH*HEADS), block 256 (8 waves).
// Each wave owns a 16-query tile. Scores S^T = K*Q^T (lane pair per query
// column -> softmax needs one shfl_xor(16)). K/V fragments are register
// double-buffered: loads for kb+32 are issued before the kb WMMAs+softmax so
// s_wait_loadcnt overlaps VALU work instead of stalling the matrix pipe.
// ---------------------------------------------------------------------------
__global__ void __launch_bounds__(256)
flash_attn(const __bf16* __restrict__ Q, const __bf16* __restrict__ Kc,
           const __bf16* __restrict__ Vt, __bf16* __restrict__ O) {
    const int lane = threadIdx.x & 31;
    const int wave = threadIdx.x >> 5;
    const int hi   = lane >> 4;
    const int lid  = lane & 15;
    const int bh   = blockIdx.y;
    const int b    = bh >> 4, h = bh & 15;
    const int qbase = (blockIdx.x * 8 + wave) * 16;

    bf16x16 bq[2];
    {
        const __bf16* qrow =
            Q + (size_t)(b * SEQ + qbase + lid) * D_MODEL + h * DK;
        bq[0] = *(const bf16x16*)(qrow + hi * 16);
        bq[1] = *(const bf16x16*)(qrow + 32 + hi * 16);
    }

    auto load_k = [&](int kb, bf16x8 (&kx)[2][2][2]) {
        #pragma unroll
        for (int t = 0; t < 2; ++t) {
            int kr = kb + t * 16 + lid;
            if (kr > SEQ - 1) kr = SEQ - 1;           // clamp (masked anyway)
            const __bf16* krow =
                Kc + (size_t)(b * SEQ + kr) * D_MODEL + h * DK;
            #pragma unroll
            for (int c = 0; c < 2; ++c) {
                kx[t][c][0] = *(const bf16x8*)(krow + c * 32 + hi * 16);
                kx[t][c][1] = *(const bf16x8*)(krow + c * 32 + hi * 16 + 8);
            }
        }
    };
    auto load_v = [&](int kb, bf16x8 (&vx)[4][2]) {
        #pragma unroll
        for (int j = 0; j < 4; ++j) {
            const __bf16* vrow = Vt + ((size_t)bh * 64 + j * 16 + lid) * SEQ;
            int c1 = kb + hi * 16;     if (c1 > SEQ - 8) c1 = SEQ - 8;
            int c2 = kb + hi * 16 + 8; if (c2 > SEQ - 8) c2 = SEQ - 8;
            vx[j][0] = *(const bf16x8*)(vrow + c1);
            vx[j][1] = *(const bf16x8*)(vrow + c2);
        }
    };

    f32x8 o[4] = {};
    float m = -3.0e38f, l = 0.f;
    const float scale = 0.125f;            // 1/sqrt(64)
    const int   kLimit = qbase + 16;       // causal coverage for this tile
    const int   q = qbase + lid;           // this lane's query column

    bf16x8 kx[2][2][2], vx[4][2];
    load_k(0, kx);
    load_v(0, vx);

    for (int kb = 0; kb < kLimit; kb += 32) {
        const bool more = (kb + 32) < kLimit;   // wave-uniform
        bf16x8 nk[2][2][2], nv[4][2];
        if (more) { load_k(kb + 32, nk); load_v(kb + 32, nv); }

        // ---- scores S^T (two 16-k-row tiles, dk=64 as 2x K=32 WMMA)
        f32x8 sT[2];
        #pragma unroll
        for (int t = 0; t < 2; ++t) {
            f32x8 accT = {};
            #pragma unroll
            for (int c = 0; c < 2; ++c) {
                bf16x16 ak;
                #pragma unroll
                for (int e = 0; e < 8; ++e) {
                    ak[e]     = kx[t][c][0][e];
                    ak[e + 8] = kx[t][c][1][e];
                }
                accT = __builtin_amdgcn_wmma_f32_16x16x32_bf16(
                    false, ak, false, bq[c], (short)0, accT, false, false);
            }
            sT[t] = accT;
        }

        // ---- scale + causal mask + online softmax
        float p0[8], p1[8];
        float mloc = -3.0e38f;
        #pragma unroll
        for (int r = 0; r < 8; ++r) {
            int k0i = kb + r + 8 * hi;
            int k1i = kb + 16 + r + 8 * hi;
            float s0 = (k0i <= q) ? sT[0][r] * scale : -3.0e38f;
            float s1 = (k1i <= q) ? sT[1][r] * scale : -3.0e38f;
            p0[r] = s0; p1[r] = s1;
            mloc = fmaxf(mloc, fmaxf(s0, s1));
        }
        mloc = fmaxf(mloc, __shfl_xor(mloc, 16, 32));
        float mnew  = fmaxf(m, mloc);
        float alpha = __expf(m - mnew);
        float lsum = 0.f;
        #pragma unroll
        for (int r = 0; r < 8; ++r) {
            p0[r] = __expf(p0[r] - mnew);
            p1[r] = __expf(p1[r] - mnew);
            lsum += p0[r] + p1[r];
        }
        l = l * alpha + lsum + __shfl_xor(lsum, 16, 32);
        m = mnew;
        #pragma unroll
        for (int j = 0; j < 4; ++j)
            #pragma unroll
            for (int r = 0; r < 8; ++r) o[j][r] *= alpha;

        // ---- repack P^T (C layout) into 32x16 B fragment via lane pairing
        bf16x16 bp;
        #pragma unroll
        for (int r = 0; r < 8; ++r) {
            float o0 = __shfl_xor(p0[r], 16, 32);
            float o1 = __shfl_xor(p1[r], 16, 32);
            float e_lo = (hi == 0) ? p0[r] : o1;
            float e_hi = (hi == 0) ? o0 : p1[r];
            bp[r]     = (__bf16)e_lo;
            bp[r + 8] = (__bf16)e_hi;
        }

        // ---- O^T += V^T * P^T  (four dv tiles of 16)
        #pragma unroll
        for (int j = 0; j < 4; ++j) {
            bf16x16 av;
            #pragma unroll
            for (int e = 0; e < 8; ++e) {
                av[e]     = vx[j][0][e];
                av[e + 8] = vx[j][1][e];
            }
            o[j] = __builtin_amdgcn_wmma_f32_16x16x32_bf16(
                false, av, false, bp, (short)0, o[j], false, false);
        }

        if (more) {
            #pragma unroll
            for (int t = 0; t < 2; ++t)
                #pragma unroll
                for (int c = 0; c < 2; ++c) {
                    kx[t][c][0] = nk[t][c][0];
                    kx[t][c][1] = nk[t][c][1];
                }
            #pragma unroll
            for (int j = 0; j < 4; ++j) {
                vx[j][0] = nv[j][0];
                vx[j][1] = nv[j][1];
            }
        }
    }

    // ---- normalize and store (lane holds q=lid, dv = j*16 + hi*8 + r)
    float invl = 1.f / l;
    #pragma unroll
    for (int j = 0; j < 4; ++j) {
        bf16x8 ov;
        #pragma unroll
        for (int r = 0; r < 8; ++r) ov[r] = (__bf16)(o[j][r] * invl);
        __bf16* orow = O + (size_t)(b * SEQ + qbase + lid) * D_MODEL
                         + h * DK + j * 16 + hi * 8;
        *(bf16x8*)orow = ov;
    }
}

// ---------------------------------------------------------------------------
// out = LayerNorm(X + Dlt) * gamma + beta   (unbiased var, ddof=1)
// ---------------------------------------------------------------------------
__global__ void __launch_bounds__(256)
residual_layernorm(const float* __restrict__ X, const float* __restrict__ Dlt,
                   const float* __restrict__ gamma, const float* __restrict__ beta,
                   float* __restrict__ outF, __bf16* __restrict__ outB) {
    __shared__ float s_sum[256], s_sq[256];
    const int row = blockIdx.x;
    const float* xr = X   + (size_t)row * D_MODEL;
    const float* dr = Dlt + (size_t)row * D_MODEL;

    float v[4];
    float sum = 0.f, sq = 0.f;
    #pragma unroll
    for (int i = 0; i < 4; ++i) {
        int c = threadIdx.x + i * 256;
        v[i] = xr[c] + dr[c];
        sum += v[i];
        sq  += v[i] * v[i];
    }
    s_sum[threadIdx.x] = sum;
    s_sq[threadIdx.x]  = sq;
    __syncthreads();
    for (int off = 128; off > 0; off >>= 1) {
        if (threadIdx.x < off) {
            s_sum[threadIdx.x] += s_sum[threadIdx.x + off];
            s_sq[threadIdx.x]  += s_sq[threadIdx.x + off];
        }
        __syncthreads();
    }
    float mean = s_sum[0] * (1.f / D_MODEL);
    float var  = (s_sq[0] - (float)D_MODEL * mean * mean) * (1.f / (D_MODEL - 1));
    float rstd = rsqrtf(var + LN_EPS);

    float* of = outF + (size_t)row * D_MODEL;
    #pragma unroll
    for (int i = 0; i < 4; ++i) {
        int c = threadIdx.x + i * 256;
        float y = gamma[c] * (v[i] - mean) * rstd + beta[c];
        of[c] = y;
        if (outB) outB[(size_t)row * D_MODEL + c] = (__bf16)y;
    }
}

// ---------------------------------------------------------------------------
// Host-side orchestration
// ---------------------------------------------------------------------------
extern "C" void kernel_launch(void* const* d_in, const int* in_sizes, int n_in,
                              void* d_out, int out_size, void* d_ws, size_t ws_size,
                              hipStream_t stream) {
    const float* x  = (const float*)d_in[0];
    // d_in[1] is the tril mask; causality is implemented analytically.
    const float* Wq = (const float*)d_in[2];
    const float* Wk = (const float*)d_in[3];
    const float* Wv = (const float*)d_in[4];
    const float* Wo = (const float*)d_in[5];
    const float* W1 = (const float*)d_in[6];
    const float* W2 = (const float*)d_in[7];
    const float* g1 = (const float*)d_in[8];
    const float* b1 = (const float*)d_in[9];
    const float* g2 = (const float*)d_in[10];
    const float* b2 = (const float*)d_in[11];

    char* ws = (char*)d_ws;
    const size_t MB = 1ull << 20;
    __bf16* xb    = (__bf16*)(ws +   0 * MB);   //  8 MB
    __bf16* Wqb   = (__bf16*)(ws +   8 * MB);   //  2 MB
    __bf16* Wkb   = (__bf16*)(ws +  10 * MB);   //  2 MB
    __bf16* Wvb   = (__bf16*)(ws +  12 * MB);   //  2 MB
    __bf16* Wob   = (__bf16*)(ws +  14 * MB);   //  2 MB
    __bf16* W1b   = (__bf16*)(ws +  16 * MB);   //  8 MB
    __bf16* W2b   = (__bf16*)(ws +  24 * MB);   //  8 MB
    __bf16* Qbuf  = (__bf16*)(ws +  32 * MB);   //  8 MB
    __bf16* Kbuf  = (__bf16*)(ws +  40 * MB);   //  8 MB
    __bf16* Vbuf  = (__bf16*)(ws +  48 * MB);   //  8 MB
    __bf16* Vt    = (__bf16*)(ws +  56 * MB);   //  8 MB
    __bf16* attnb = (__bf16*)(ws +  64 * MB);   //  8 MB
    float*  proj  = (float*) (ws +  72 * MB);   // 16 MB
    float*  h1f   = (float*) (ws +  88 * MB);   // 16 MB
    __bf16* h1b   = (__bf16*)(ws + 104 * MB);   //  8 MB
    __bf16* ff1b  = (__bf16*)(ws + 112 * MB);   // 32 MB
    float*  ff2   = (float*) (ws + 144 * MB);   // 16 MB  -> 160 MB total

    const dim3 blk(256);
    auto cvt = [&](const float* src, __bf16* dst, int n) {
        cvt_f32_bf16<<<dim3((n + 255) / 256), blk, 0, stream>>>(src, dst, n);
    };

    cvt(x,  xb,  ROWS * D_MODEL);
    cvt(Wq, Wqb, D_MODEL * D_MODEL);
    cvt(Wk, Wkb, D_MODEL * D_MODEL);
    cvt(Wv, Wvb, D_MODEL * D_MODEL);
    cvt(Wo, Wob, D_MODEL * D_MODEL);
    cvt(W1, W1b, D_FF * D_MODEL);
    cvt(W2, W2b, D_MODEL * D_FF);

    dim3 gProj(D_MODEL / 256, ROWS / 32);   // (4, 128)
    gemm_bf16_nt<1><<<gProj, blk, 0, stream>>>(xb, Wqb, Qbuf, ROWS, D_MODEL, D_MODEL);
    gemm_bf16_nt<1><<<gProj, blk, 0, stream>>>(xb, Wkb, Kbuf, ROWS, D_MODEL, D_MODEL);
    gemm_bf16_nt<1><<<gProj, blk, 0, stream>>>(xb, Wvb, Vbuf, ROWS, D_MODEL, D_MODEL);

    {
        int total = BATCH * HEADS * DK * SEQ;   // 4,194,304
        transpose_v<<<dim3(total / 256), blk, 0, stream>>>(Vbuf, Vt, total);
    }

    flash_attn<<<dim3(SEQ / 128, BATCH * HEADS), blk, 0, stream>>>(
        Qbuf, Kbuf, Vt, attnb);

    gemm_bf16_nt<0><<<gProj, blk, 0, stream>>>(attnb, Wob, proj, ROWS, D_MODEL, D_MODEL);
    residual_layernorm<<<dim3(ROWS), blk, 0, stream>>>(x, proj, g1, b1, h1f, h1b);

    gemm_bf16_nt<2><<<dim3(D_FF / 256, ROWS / 32), blk, 0, stream>>>(
        h1b, W1b, ff1b, ROWS, D_FF, D_MODEL);
    gemm_bf16_nt<0><<<dim3(D_MODEL / 256, ROWS / 32), blk, 0, stream>>>(
        ff1b, W2b, ff2, ROWS, D_MODEL, D_FF);

    residual_layernorm<<<dim3(ROWS), blk, 0, stream>>>(
        h1f, ff2, g2, b2, (float*)d_out, nullptr);
}
